// DotProductAttention_30631706755291
// MI455X (gfx1250) — matrix operand order repaired and checked by reading, not compiled, not run on hardware
//
#include <hip/hip_runtime.h>

// Fused masked-norm-softmax attention for MI455X (gfx1250, wave32, WMMA).
//
//   out = softmax(mask(normalize(QK^T - rowmin), -1e6)) @ V
//
// v4 changes vs v3:
//  * QTILE 16 -> 32 with 16-wave (512-thread) workgroups: waves w and w+8
//    process two q-row blocks with identical K/V partitioning, so the
//    second block's fragment fetches hit the WGP cache -> L2 traffic
//    halves (4 GB -> 2 GB) while per-wave VGPR footprint is unchanged.
//  * 4 waves/SIMD from one WG (was 2) for better latency hiding.
//  * P panel: 32 rows x 512 cols (33 KB), 4 chunks.
// Retained: f16 pre-pass (K row-major, V transposed [dv][k]), pass-A loop
// interchange (16 independent WMMA chains), register-resident packed-f16 S.

typedef _Float16 half16 __attribute__((ext_vector_type(16)));
typedef _Float16 half8  __attribute__((ext_vector_type(8)));
typedef float    float8 __attribute__((ext_vector_type(8)));

constexpr int BATCH = 8;
constexpr int QLEN  = 2048;
constexpr int KLEN  = 2048;
constexpr int DDIM  = 512;
constexpr int DVDIM = 512;
constexpr int QTILE = 32;     // q rows per workgroup (2 blocks of 16)
constexpr int NWAVE = 16;     // waves per workgroup (512 threads, wave32)
constexpr int NWQB  = 8;      // waves per q-block
constexpr int CHUNK = 512;    // K columns of P staged in LDS per chunk
constexpr int PPAD  = 8;      // f16 pad -> row stride 260 dw = 4 mod 64 banks

__device__ inline float red_sum16(float v) {
  v += __shfl_xor(v, 1, 32);
  v += __shfl_xor(v, 2, 32);
  v += __shfl_xor(v, 4, 32);
  v += __shfl_xor(v, 8, 32);
  return v;
}
__device__ inline float red_min16(float v) {
  v = fminf(v, __shfl_xor(v, 1, 32));
  v = fminf(v, __shfl_xor(v, 2, 32));
  v = fminf(v, __shfl_xor(v, 4, 32));
  v = fminf(v, __shfl_xor(v, 8, 32));
  return v;
}
__device__ inline float8 zero8() {
  float8 z;
#pragma unroll
  for (int i = 0; i < 8; ++i) z[i] = 0.0f;
  return z;
}

// ---------------- pre-pass 1: K fp32 -> f16, same layout -----------------
__global__ __launch_bounds__(256)
void cvt_k_kernel(const float* __restrict__ src, _Float16* __restrict__ dst) {
  const size_t i = ((size_t)blockIdx.x * 256 + threadIdx.x) * 8;
  const float4* sp = reinterpret_cast<const float4*>(src + i);
  float4 a = sp[0], b = sp[1];
  union { half8 h; uint4 q; } o;
  o.h[0] = (_Float16)a.x; o.h[1] = (_Float16)a.y;
  o.h[2] = (_Float16)a.z; o.h[3] = (_Float16)a.w;
  o.h[4] = (_Float16)b.x; o.h[5] = (_Float16)b.y;
  o.h[6] = (_Float16)b.z; o.h[7] = (_Float16)b.w;
  *reinterpret_cast<uint4*>(dst + i) = o.q;
}

// ------- pre-pass 2: V fp32 [k][dv] -> f16 transposed [dv][k] ------------
__global__ __launch_bounds__(256)
void cvt_v_kernel(const float* __restrict__ src, _Float16* __restrict__ dst) {
  __shared__ _Float16 tile[64][72];          // 64x64 tile, padded rows
  const int b  = blockIdx.x >> 8;            // 256 tiles per batch
  const int kt = (blockIdx.x & 255) >> 3;    // 32 k-tiles
  const int dt = blockIdx.x & 7;             // 8 dv-tiles
  const int t  = threadIdx.x;

  {  // load + convert: each thread handles 16 consecutive dv of one k row
    const int kr  = t >> 2;
    const int dcb = (t & 3) << 4;
    const float* sp =
        src + ((size_t)(b * KLEN + kt * 64 + kr)) * DVDIM + dt * 64 + dcb;
    const float4* sp4 = reinterpret_cast<const float4*>(sp);
    float4 x0 = sp4[0], x1 = sp4[1], x2 = sp4[2], x3 = sp4[3];
    _Float16* tp = &tile[kr][dcb];
    tp[0]  = (_Float16)x0.x; tp[1]  = (_Float16)x0.y;
    tp[2]  = (_Float16)x0.z; tp[3]  = (_Float16)x0.w;
    tp[4]  = (_Float16)x1.x; tp[5]  = (_Float16)x1.y;
    tp[6]  = (_Float16)x1.z; tp[7]  = (_Float16)x1.w;
    tp[8]  = (_Float16)x2.x; tp[9]  = (_Float16)x2.y;
    tp[10] = (_Float16)x2.z; tp[11] = (_Float16)x2.w;
    tp[12] = (_Float16)x3.x; tp[13] = (_Float16)x3.y;
    tp[14] = (_Float16)x3.z; tp[15] = (_Float16)x3.w;
  }
  __syncthreads();
  {  // transpose out: each thread writes 16 consecutive k of one dv row
    const int dvr = t >> 2;
    const int kcb = (t & 3) << 4;
    union { half16 h; uint4 q[2]; } o;
#pragma unroll
    for (int j = 0; j < 16; ++j) o.h[j] = tile[kcb + j][dvr];
    _Float16* dp =
        dst + ((size_t)(b * DVDIM + dt * 64 + dvr)) * KLEN + kt * 64 + kcb;
    uint4* dp4 = reinterpret_cast<uint4*>(dp);
    dp4[0] = o.q[0];
    dp4[1] = o.q[1];
  }
}

// --------------------------- main kernel ---------------------------------
__global__ __launch_bounds__(NWAVE * 32)
void fused_attn_kernel(const float* __restrict__ Qm,
                       const _Float16* __restrict__ Kh,   // f16 [b][k][d]
                       const _Float16* __restrict__ Vt,   // f16 [b][dv][k]
                       const long long* __restrict__ VL,
                       float* __restrict__ Out) {
  __shared__ _Float16 panel[QTILE][CHUNK + PPAD];   // staged P (f16), 33 KB
  __shared__ float sMinP[NWAVE][16];
  __shared__ float sSumP[NWAVE][16];
  __shared__ float sSqP [NWAVE][16];
  __shared__ float sRsP [NWAVE][16];
  __shared__ float sMinF[QTILE];
  __shared__ float sInvD[QTILE];
  __shared__ float sRsI [QTILE];

  const int tid     = threadIdx.x;
  const int lane    = tid & 31;
  const int wv      = tid >> 5;       // 0..15
  const int wv8     = wv & 7;         // wave id within q-block
  const int qblk    = wv >> 3;        // which 16-row q-block (0/1)
  const int laneN   = lane & 15;
  const int halfSel = lane >> 4;
  const int halfOff = halfSel << 3;

  const int b    = blockIdx.x >> 6;          // 64 q-tiles of 32 per batch
  const int q0wg = (blockIdx.x & 63) << 5;   // first q row of this WG
  const int q0   = q0wg + qblk * 16;         // first q row of this wave

  const float*    Qp  = Qm + ((size_t)b * QLEN + q0) * DDIM;
  const _Float16* Kpb = Kh + (size_t)b * KLEN * DDIM;
  const _Float16* Vpb = Vt + (size_t)b * DVDIM * KLEN;

  const int   vl    = (int)VL[b];
  const float fillp = (vl > 0) ? 0.0f : 1.0f;  // vl==0: jax softmax -> uniform

  // ======== Pass A: S = Q K^T, D-chunk OUTER / k-tile INNER =============
  // 16 independent accumulators -> 16 parallel WMMA chains per D-chunk;
  // only one 8-VGPR Q fragment is live at a time.  Waves w and w+8 issue
  // identical K-fragment addresses -> second hit lands in WGP cache.
  float8 acc[16];
#pragma unroll
  for (int i = 0; i < 16; ++i) acc[i] = zero8();

  const float* qrow = Qp + (size_t)laneN * DDIM;
  const _Float16* krow0 =
      Kpb + (size_t)(wv8 * 16 + laneN) * DDIM + halfSel * 16;
  constexpr size_t KT_STRIDE = (size_t)NWQB * 16 * DDIM;  // 8 tiles apart

  for (int c = 0; c < 16; ++c) {
    // Q A-fragment for this D-chunk (fp32 -> f16, once per chunk)
    const int d0 = c * 32 + halfSel * 16;
    float4 x0 = *reinterpret_cast<const float4*>(qrow + d0);
    float4 x1 = *reinterpret_cast<const float4*>(qrow + d0 + 4);
    float4 y0 = *reinterpret_cast<const float4*>(qrow + d0 + 8);
    float4 y1 = *reinterpret_cast<const float4*>(qrow + d0 + 12);
    half16 a;
    a[0]  = (_Float16)x0.x; a[1]  = (_Float16)x0.y;
    a[2]  = (_Float16)x0.z; a[3]  = (_Float16)x0.w;
    a[4]  = (_Float16)x1.x; a[5]  = (_Float16)x1.y;
    a[6]  = (_Float16)x1.z; a[7]  = (_Float16)x1.w;
    a[8]  = (_Float16)y0.x; a[9]  = (_Float16)y0.y;
    a[10] = (_Float16)y0.z; a[11] = (_Float16)y0.w;
    a[12] = (_Float16)y1.x; a[13] = (_Float16)y1.y;
    a[14] = (_Float16)y1.z; a[15] = (_Float16)y1.w;

    const _Float16* kp = krow0 + c * 32;
#pragma unroll
    for (int i = 0; i < 16; ++i) {          // kt = wv8 + 8*i (independent)
      union { half16 h; uint4 q[2]; } f;
      const uint4* kq = reinterpret_cast<const uint4*>(kp + i * KT_STRIDE);
      f.q[0] = kq[0];
      f.q[1] = kq[1];
      acc[i] = __builtin_amdgcn_wmma_f32_16x16x32_f16(
          false, a, false, f.h, (short)0, acc[i], false, false);
    }
  }

  // ---- stats + pack S to register-resident f16 (64 VGPRs) ----
  half8 sS[16];
  float mn[8], sm[8], sq[8];
#pragma unroll
  for (int r = 0; r < 8; ++r) { mn[r] = 3.4e38f; sm[r] = 0.f; sq[r] = 0.f; }
#pragma unroll
  for (int i = 0; i < 16; ++i) {
    half8 pk;
#pragma unroll
    for (int r = 0; r < 8; ++r) {
      float v = acc[i][r];
      mn[r] = fminf(mn[r], v);
      sm[r] += v;
      sq[r] += v * v;
      pk[r] = (_Float16)v;
    }
    sS[i] = pk;
  }

#pragma unroll
  for (int r = 0; r < 8; ++r) {
    mn[r] = red_min16(mn[r]);
    sm[r] = red_sum16(sm[r]);
    sq[r] = red_sum16(sq[r]);
  }
  if (laneN == 0) {
#pragma unroll
    for (int r = 0; r < 8; ++r) {
      sMinP[wv][halfOff + r] = mn[r];
      sSumP[wv][halfOff + r] = sm[r];
      sSqP [wv][halfOff + r] = sq[r];
    }
  }
  __syncthreads();
  if (tid < QTILE) {            // one thread per q-row (32 rows)
    const int qb = tid >> 4, rl = tid & 15;
    float m = 3.4e38f, s = 0.f, q = 0.f;
#pragma unroll
    for (int w8 = 0; w8 < NWQB; ++w8) {
      const int w = qb * NWQB + w8;
      m = fminf(m, sMinP[w][rl]);
      s += sSumP[w][rl];
      q += sSqP[w][rl];
    }
    float var = q - 2.0f * m * s + (float)KLEN * m * m;  // sum((s-min)^2)
    var = fmaxf(var, 0.0f);
    float den = fmaxf(sqrtf(var), 1e-12f);
    sMinF[tid] = m;
    sInvD[tid] = 1.0f / den;
  }
  __syncthreads();

  float mnR[8], idR[8];
#pragma unroll
  for (int r = 0; r < 8; ++r) {
    mnR[r] = sMinF[qblk * 16 + halfOff + r];
    idR[r] = sInvD[qblk * 16 + halfOff + r];
  }

  // ======== Pass B: P from registers -> LDS panel; O += P @ V ===========
  float rsum[8];
#pragma unroll
  for (int r = 0; r < 8; ++r) rsum[r] = 0.f;
  float8 oacc[4];
#pragma unroll
  for (int t = 0; t < 4; ++t) oacc[t] = zero8();

  const int dvb = wv8 * (DVDIM / NWQB);   // 64-wide DV slice per q-block wave
  const _Float16* vbase0 = Vpb + (size_t)(dvb + 0 * 16 + laneN) * KLEN;
  const _Float16* vbase1 = Vpb + (size_t)(dvb + 1 * 16 + laneN) * KLEN;
  const _Float16* vbase2 = Vpb + (size_t)(dvb + 2 * 16 + laneN) * KLEN;
  const _Float16* vbase3 = Vpb + (size_t)(dvb + 3 * 16 + laneN) * KLEN;
  const _Float16* vbase[4] = {vbase0, vbase1, vbase2, vbase3};

#pragma unroll
  for (int ch = 0; ch < KLEN / CHUNK; ++ch) {      // 4 chunks of 512
    // ---- transform this wave's register-held S tiles into P (no GEMM) ----
#pragma unroll
    for (int ii = 0; ii < 4; ++ii) {       // 4 tiles per wave per chunk
      const int i   = ch * 4 + ii;
      const int kt  = wv8 + i * NWQB;      // global k-tile (= 32*ch + ...)
      const int ktl = kt - ch * 32;        // tile within chunk (0..31)
      const int  colg   = kt * 16 + laneN;
      const bool masked = (colg >= vl);
      const half8 pk = sS[i];
#pragma unroll
      for (int r = 0; r < 8; ++r) {
        float x = ((float)pk[r] - mnR[r]) * idR[r];   // in [0,1]
        float p = masked ? fillp : __expf(x);
        rsum[r] += p;
        panel[qblk * 16 + halfOff + r][ktl * 16 + laneN] = (_Float16)p;
      }
    }
    __syncthreads();
    // ---- O += P_chunk @ V_chunk; V B-fragment = contiguous Vt slice ----
#pragma unroll 2
    for (int kg = 0; kg < CHUNK / 32; ++kg) {      // 16 iterations
      const int pbase = kg * 32 + halfSel * 16;
      union { half16 h; uint4 q[2]; } pf;
      pf.q[0] = *reinterpret_cast<const uint4*>(&panel[qblk * 16 + laneN][pbase]);
      pf.q[1] = *reinterpret_cast<const uint4*>(&panel[qblk * 16 + laneN][pbase + 8]);
      const int kvb = ch * CHUNK + kg * 32 + halfSel * 16;
#pragma unroll
      for (int t = 0; t < 4; ++t) {
        const uint4* vp = reinterpret_cast<const uint4*>(vbase[t] + kvb);
        union { half16 h; uint4 q[2]; } bv;
        bv.q[0] = vp[0];
        bv.q[1] = vp[1];
        oacc[t] = __builtin_amdgcn_wmma_f32_16x16x32_f16(
            false, pf.h, false, bv.h, (short)0, oacc[t], false, false);
      }
    }
    __syncthreads();   // panel reused next chunk
  }

  // ---- reduce row exp-sums across lanes and waves ----
#pragma unroll
  for (int r = 0; r < 8; ++r) rsum[r] = red_sum16(rsum[r]);
  if (laneN == 0) {
#pragma unroll
    for (int r = 0; r < 8; ++r) sRsP[wv][halfOff + r] = rsum[r];
  }
  __syncthreads();
  if (tid < QTILE) {
    const int qb = tid >> 4, rl = tid & 15;
    float s = 0.f;
#pragma unroll
    for (int w8 = 0; w8 < NWQB; ++w8) s += sRsP[qb * NWQB + w8][rl];
    sRsI[tid] = 1.0f / s;   // <= 1 when vl>0; == 1/K when vl==0
  }
  __syncthreads();

  float irs[8];
#pragma unroll
  for (int r = 0; r < 8; ++r) irs[r] = sRsI[qblk * 16 + halfOff + r];

  float* orow = Out + ((size_t)b * QLEN + q0) * DVDIM;
#pragma unroll
  for (int t = 0; t < 4; ++t) {
#pragma unroll
    for (int r = 0; r < 8; ++r) {
      const int row = halfOff + r;
      orow[(size_t)row * DVDIM + dvb + t * 16 + laneN] = oacc[t][r] * irs[r];
    }
  }
}

extern "C" void kernel_launch(void* const* d_in, const int* in_sizes, int n_in,
                              void* d_out, int out_size, void* d_ws, size_t ws_size,
                              hipStream_t stream) {
  (void)in_sizes; (void)n_in; (void)out_size; (void)ws_size;
  const float*     Qm = (const float*)d_in[0];
  const float*     Km = (const float*)d_in[1];
  const float*     Vm = (const float*)d_in[2];
  const long long* VL = (const long long*)d_in[3];   // int64 valid_length
  float*           Out = (float*)d_out;

  // Workspace: Kh (f16, 16 MiB) then Vt (f16 transposed, 16 MiB).
  constexpr size_t KH_BYTES = (size_t)BATCH * KLEN * DDIM * sizeof(_Float16);
  _Float16* Kh = (_Float16*)d_ws;
  _Float16* Vt = (_Float16*)((char*)d_ws + KH_BYTES);

  {  // K fp32 -> f16 (8 elems/thread)
    const int blocks = (int)((size_t)BATCH * KLEN * DDIM / (8 * 256));
    cvt_k_kernel<<<dim3(blocks), dim3(256), 0, stream>>>(Km, Kh);
  }
  {  // V fp32 [k][dv] -> f16 [dv][k] (64x64 tiles)
    const int blocks = BATCH * (KLEN / 64) * (DVDIM / 64);
    cvt_v_kernel<<<dim3(blocks), dim3(256), 0, stream>>>(Vm, Vt);
  }
  fused_attn_kernel<<<dim3(BATCH * (QLEN / QTILE)), dim3(NWAVE * 32), 0,
                      stream>>>(Qm, Kh, Vt, VL, Out);
}
